// RelativeMultiHeadAttention_3298534884090
// MI455X (gfx1250) — compile-verified, matrix operand order checked by reading, not run on hardware
//
#include <hip/hip_runtime.h>

#define DMODEL 1024
#define NHEADS 16
#define DHEAD  64
#define BATCH  2
#define QLEN   1024
#define MEMLEN 1024
#define KLEN   2048

typedef __bf16 bf16_t;
typedef bf16_t v16bf __attribute__((ext_vector_type(16)));
typedef float  v8f   __attribute__((ext_vector_type(8)));

#define V8F_ZERO {0.f,0.f,0.f,0.f,0.f,0.f,0.f,0.f}

// Attention LDS: 16 x KLEN fp32 scores + 4x16x16 fp32 P@V partials
#define ATT_LDS_BYTES (16 * KLEN * 4 + 4 * 16 * 16 * 4)

static __device__ __forceinline__ bf16_t f2bf(float f) {
  unsigned u = __float_as_uint(f);
  u += 0x7fffu + ((u >> 16) & 1u);            // round-to-nearest-even
  unsigned short s = (unsigned short)(u >> 16);
  return __builtin_bit_cast(bf16_t, s);
}

union Frag16 { v16bf v; uint4 q[2]; };

// Two contiguous 16-byte loads per lane from a precomputed lane base pointer,
// matching the CDNA5 16-bit A-matrix VGPR layout (also B when stored N x K).
static __device__ __forceinline__ v16bf load_frag(const bf16_t* __restrict__ laneBase,
                                                  int k0) {
  Frag16 f;
  f.q[0] = *reinterpret_cast<const uint4*>(laneBase + k0);
  f.q[1] = *reinterpret_cast<const uint4*>(laneBase + k0 + 16);
  return f.v;
}

// Generic fragment load (row-major; ld constant-folds after inlining).
static __device__ __forceinline__ v16bf load_frag_k32(const bf16_t* __restrict__ src,
                                                      int ld, int row0, int k0, int lane) {
  int m  = lane & 15;
  int kh = lane >> 4;
  return load_frag(src + (size_t)(row0 + m) * ld + kh * 8, k0);
}

// A fragment from fp32 scores in LDS, converting to bf16 on the fly.
static __device__ __forceinline__ v16bf load_a_lds_f32(const float* sc, int k0, int lane) {
  int m  = lane & 15;
  int kh = lane >> 4;
  Frag16 f;
  const float* row = sc + (size_t)m * KLEN;
#pragma unroll
  for (int e = 0; e < 8; ++e) {
    f.v[e]     = f2bf(row[k0 + kh * 8 + e]);
    f.v[8 + e] = f2bf(row[k0 + 16 + kh * 8 + e]);
  }
  return f.v;
}

static __device__ __forceinline__ v8f wmma_bf16(v16bf a, v16bf b, v8f c) {
  return __builtin_amdgcn_wmma_f32_16x16x32_bf16(false, a, false, b, (short)0, c, false, false);
}

// ---------------------------------------------------------------------------
// Elementwise prep kernels
// ---------------------------------------------------------------------------
__global__ void k_build_cat(const float* __restrict__ mem, const float* __restrict__ h,
                            bf16_t* __restrict__ catb, bf16_t* __restrict__ hb) {
  size_t i = (size_t)blockIdx.x * blockDim.x + threadIdx.x;
  size_t total = (size_t)BATCH * KLEN * DMODEL;
  if (i >= total) return;
  size_t b = i / ((size_t)KLEN * DMODEL);
  size_t rem = i % ((size_t)KLEN * DMODEL);
  size_t t = rem / DMODEL;
  size_t d = rem % DMODEL;
  float v;
  if (t < MEMLEN) {
    v = mem[b * (size_t)MEMLEN * DMODEL + t * DMODEL + d];
  } else {
    size_t th = t - MEMLEN;
    v = h[b * (size_t)QLEN * DMODEL + th * DMODEL + d];
    hb[b * (size_t)QLEN * DMODEL + th * DMODEL + d] = f2bf(v);
  }
  catb[i] = f2bf(v);
}

__global__ void k_cast_bf16(const float* __restrict__ src, bf16_t* __restrict__ dst,
                            unsigned long long n) {
  size_t i = (size_t)blockIdx.x * blockDim.x + threadIdx.x;
  if (i < n) dst[i] = f2bf(src[i]);
}

__global__ void k_bias_cast(const float* __restrict__ qf, const float* __restrict__ rwb,
                            const float* __restrict__ rrb, bf16_t* __restrict__ qwb,
                            bf16_t* __restrict__ qrb) {
  size_t i = (size_t)blockIdx.x * blockDim.x + threadIdx.x;
  size_t total = (size_t)BATCH * QLEN * DMODEL;
  if (i >= total) return;
  int col = (int)(i % DMODEL);   // col = head*64 + dh, biases are (16,64) flat
  float q = qf[i];
  qwb[i] = f2bf(q + rwb[col]);
  qrb[i] = f2bf(q + rrb[col]);
}

// Batched tiled transpose+cast: per z, dst(cols x rows) = src(rows x cols)^T.
// blockDim (32, 8); grid (rows/32, cols/32, nbatch).
__global__ void k_transpose_cast(const float* __restrict__ src, bf16_t* __restrict__ dst,
                                 int rows, int cols) {
  __shared__ float tile[32][33];
  size_t zoff = (size_t)blockIdx.z * rows * cols;
  const float* s = src + zoff;
  bf16_t* d = dst + zoff;
  int bx = blockIdx.x * 32;  // src row block
  int by = blockIdx.y * 32;  // src col block
  int tx = threadIdx.x, ty = threadIdx.y;
#pragma unroll
  for (int r = 0; r < 32; r += 8)
    tile[ty + r][tx] = s[(size_t)(bx + ty + r) * cols + by + tx];
  __syncthreads();
#pragma unroll
  for (int r = 0; r < 32; r += 8)
    d[(size_t)(by + ty + r) * rows + bx + tx] = f2bf(tile[tx][ty + r]);
}

// ---------------------------------------------------------------------------
// bf16 WMMA GEMM: C(MxN, fp32) = A(MxK) x Bt(NxK)^T, LDA/LDB compile-time so
// the four B-row fragments are addressed via instruction immediates off one
// pointer. One wave per 16x64 C strip; K step 32; software-pipelined so
// next-step global_load_b128 overlap current WMMAs.
// ---------------------------------------------------------------------------
template <int LDA, int LDB>
__global__ void gemm_bf16_wmma(const bf16_t* __restrict__ A,
                               const bf16_t* __restrict__ Bt,
                               float* __restrict__ C, int ldc,
                               int M, int N, int K) {
  int lane = threadIdx.x & 31;
  int wave = (int)((blockIdx.x * blockDim.x + threadIdx.x) >> 5);
  int ntb = N >> 6;
  int total = (M >> 4) * ntb;
  if (wave >= total) return;  // wave-uniform: EXEC stays all-1s for WMMA
  int mt = wave / ntb, nt = wave % ntb;
  int row0 = mt << 4, col0 = nt << 6;

  int m = lane & 15, kh = lane >> 4;
  // Single loop-carried pointer per matrix; B rows 16/32/48 away are
  // compile-time byte offsets (32/64/96 KB) folded into load immediates.
  const bf16_t* pA = A  + (size_t)(row0 + m) * LDA + kh * 8;
  const bf16_t* pB = Bt + (size_t)(col0 + m) * LDB + kh * 8;

  v8f acc0 = V8F_ZERO, acc1 = V8F_ZERO, acc2 = V8F_ZERO, acc3 = V8F_ZERO;

  // prologue fragments (k0 = 0)
  v16bf a  = load_frag(pA, 0);
  v16bf b0 = load_frag(pB, 0);
  v16bf b1 = load_frag(pB + 16 * LDB, 0);
  v16bf b2 = load_frag(pB + 32 * LDB, 0);
  v16bf b3 = load_frag(pB + 48 * LDB, 0);

  for (int k0 = 32; k0 < K; k0 += 32) {
    if (k0 + 32 < K)
      __builtin_prefetch(pA + k0 + 32, 0, 1);
    // issue next-step loads before consuming current fragments
    v16bf an  = load_frag(pA, k0);
    v16bf b0n = load_frag(pB, k0);
    v16bf b1n = load_frag(pB + 16 * LDB, k0);
    v16bf b2n = load_frag(pB + 32 * LDB, k0);
    v16bf b3n = load_frag(pB + 48 * LDB, k0);
    acc0 = wmma_bf16(a, b0, acc0);
    acc1 = wmma_bf16(a, b1, acc1);
    acc2 = wmma_bf16(a, b2, acc2);
    acc3 = wmma_bf16(a, b3, acc3);
    a = an; b0 = b0n; b1 = b1n; b2 = b2n; b3 = b3n;
  }
  acc0 = wmma_bf16(a, b0, acc0);
  acc1 = wmma_bf16(a, b1, acc1);
  acc2 = wmma_bf16(a, b2, acc2);
  acc3 = wmma_bf16(a, b3, acc3);

  int n = lane & 15, mh = lane >> 4;
#pragma unroll
  for (int r = 0; r < 8; ++r) {
    size_t rowoff = (size_t)(row0 + r + mh * 8) * ldc + col0 + n;
    C[rowoff +  0] = acc0[r];
    C[rowoff + 16] = acc1[r];
    C[rowoff + 32] = acc2[r];
    C[rowoff + 48] = acc3[r];
  }
}

// ---------------------------------------------------------------------------
// Fused attention: per (q-tile of 16, head). Scores = (qw.K^T + shift(BD))/8,
// softmax over 2048 keys in LDS, then P@V -> attn_vec (8 waves, K split).
// ---------------------------------------------------------------------------
__global__ void attn_kernel(const bf16_t* __restrict__ qwb,
                            const bf16_t* __restrict__ kb,
                            const bf16_t* __restrict__ vtb,  // (B, DMODEL, KLEN) bf16
                            const float* __restrict__ BD,    // [head][QLEN][KLEN] this batch
                            float* __restrict__ attn_vec,    // (B*QLEN, DMODEL)
                            int b) {
  extern __shared__ float sc[];  // 16 x KLEN scores, then 4x16x16 partials
  int qt = blockIdx.x;           // 0..63
  int hh = blockIdx.y;           // 0..15
  int lane = threadIdx.x & 31;
  int w = threadIdx.x >> 5;      // 0..7
  const int col0 = hh * DHEAD;
  const int n = lane & 15, mh = lane >> 4;
  const float* BDh = BD + (size_t)hh * QLEN * KLEN;

  // ---- phase 1: AC via WMMA + shifted BD gather -> LDS scores ----
  // A fragments (qw tile) are invariant across k-tiles: hoist them.
  v16bf a0 = load_frag_k32(qwb, DMODEL, b * QLEN + qt * 16, col0,      lane);
  v16bf a1 = load_frag_k32(qwb, DMODEL, b * QLEN + qt * 16, col0 + 32, lane);
  for (int kt = w; kt < KLEN / 16; kt += 8) {
    v8f acc = V8F_ZERO;
    v16bf bk0 = load_frag_k32(kb, DMODEL, b * KLEN + kt * 16, col0,      lane);
    v16bf bk1 = load_frag_k32(kb, DMODEL, b * KLEN + kt * 16, col0 + 32, lane);
    acc = wmma_bf16(a0, bk0, acc);
    acc = wmma_bf16(a1, bk1, acc);
    // Branchless shifted-BD gather: clamp the pad case to a valid address and
    // select 0 afterwards (avoids per-element exec-mask branches around loads).
    float bdv[8];
    bool  pad[8];
#pragma unroll
    for (int r = 0; r < 8; ++r) {
      int qg = qt * 16 + r + mh * 8;
      int kg = kt * 16 + n;
      // exact reference rel_shift: flat index into zero-padded (Q, K+1)
      unsigned i = (unsigned)qg * KLEN + (unsigned)kg + QLEN;
      unsigned qp  = i / (KLEN + 1);
      unsigned kpp = i % (KLEN + 1);
      pad[r] = (kpp == 0);
      unsigned j = pad[r] ? 0u : kpp - 1u;
      bdv[r] = BDh[(size_t)qp * KLEN + j];
    }
#pragma unroll
    for (int r = 0; r < 8; ++r) {
      int ql = r + mh * 8;
      int kg = kt * 16 + n;
      float bd = pad[r] ? 0.f : bdv[r];
      sc[(size_t)ql * KLEN + kg] = (acc[r] + bd) * 0.125f;  // 1/sqrt(64)
    }
  }
  __syncthreads();

  // ---- phase 2: softmax, each wave owns 2 rows ----
  for (int rr = 2 * w; rr < 2 * w + 2; ++rr) {
    float* row = sc + (size_t)rr * KLEN;
    float mx = -3.4e38f;
    for (int j = lane; j < KLEN; j += 32) mx = fmaxf(mx, row[j]);
#pragma unroll
    for (int s = 16; s; s >>= 1) mx = fmaxf(mx, __shfl_xor(mx, s, 32));
    float sum = 0.f;
    for (int j = lane; j < KLEN; j += 32) {
      float e = __expf(row[j] - mx);
      row[j] = e;
      sum += e;
    }
#pragma unroll
    for (int s = 16; s; s >>= 1) sum += __shfl_xor(sum, s, 32);
    float inv = 1.f / sum;
    for (int j = lane; j < KLEN; j += 32) row[j] *= inv;
  }
  __syncthreads();

  // ---- phase 3: P @ V with transposed V; all 8 waves (K split in halves) ----
  int ntile = w & 3;   // 16-col output tile
  int half  = w >> 2;  // K half: [half*1024, half*1024+1024)
  const bf16_t* vtlane = vtb + (size_t)b * DMODEL * KLEN
                             + (size_t)(col0 + ntile * 16 + n) * KLEN + mh * 8;
  v8f acc = V8F_ZERO;
  int kbeg = half * (KLEN / 2), kend = kbeg + KLEN / 2;
  for (int k0 = kbeg; k0 < kend; k0 += 32) {
    v16bf a  = load_a_lds_f32(sc, k0, lane);
    v16bf bf = load_frag(vtlane, k0);
    acc = wmma_bf16(a, bf, acc);
  }
  float* part = sc + (size_t)16 * KLEN;  // 4 x 16 x 16 partials
  if (half == 1) {  // wave-uniform
#pragma unroll
    for (int r = 0; r < 8; ++r)
      part[ntile * 256 + (r + mh * 8) * 16 + n] = acc[r];
  }
  __syncthreads();
  if (half == 0) {  // wave-uniform
#pragma unroll
    for (int r = 0; r < 8; ++r) {
      float v = acc[r] + part[ntile * 256 + (r + mh * 8) * 16 + n];
      attn_vec[(size_t)(b * QLEN + qt * 16 + r + mh * 8) * DMODEL + col0 + ntile * 16 + n] = v;
    }
  }
}

// ---------------------------------------------------------------------------
// Host launcher
// ---------------------------------------------------------------------------
static inline int eblocks(size_t n) { return (int)((n + 255) / 256); }

extern "C" void kernel_launch(void* const* d_in, const int* in_sizes, int n_in,
                              void* d_out, int out_size, void* d_ws, size_t ws_size,
                              hipStream_t stream) {
  const float* h   = (const float*)d_in[0];
  const float* mem = (const float*)d_in[1];
  const float* r   = (const float*)d_in[2];
  const float* Wq  = (const float*)d_in[3];
  const float* Wk  = (const float*)d_in[4];
  const float* Wv  = (const float*)d_in[5];
  const float* Wr  = (const float*)d_in[6];
  const float* Wo  = (const float*)d_in[7];
  const float* rwb = (const float*)d_in[8];
  const float* rrb = (const float*)d_in[9];
  float* out = (float*)d_out;

  char* ws = (char*)d_ws;
  size_t off = 0;
  auto alloc = [&](size_t bytes) -> void* {
    void* p = ws + off;
    off = (off + bytes + 255) & ~(size_t)255;
    return p;
  };

  const size_t nH   = (size_t)BATCH * QLEN * DMODEL;   // 2M
  const size_t nCat = (size_t)BATCH * KLEN * DMODEL;   // 4M
  const size_t nR   = (size_t)KLEN * DMODEL;           // 2M
  const size_t nW   = (size_t)DMODEL * DMODEL;         // 1M

  bf16_t* catb = (bf16_t*)alloc(nCat * 2);
  bf16_t* hb   = (bf16_t*)alloc(nH * 2);
  bf16_t* rb   = (bf16_t*)alloc(nR * 2);
  bf16_t* WqT  = (bf16_t*)alloc(nW * 2);
  bf16_t* WkT  = (bf16_t*)alloc(nW * 2);
  bf16_t* WvT  = (bf16_t*)alloc(nW * 2);
  bf16_t* WrT  = (bf16_t*)alloc(nW * 2);
  bf16_t* WoT  = (bf16_t*)alloc(nW * 2);
  float*  qf   = (float*)alloc(nH * 4);
  float*  kf   = (float*)alloc(nCat * 4);
  float*  vf   = (float*)alloc(nCat * 4);
  float*  rhf  = (float*)alloc(nR * 4);
  bf16_t* qwb  = (bf16_t*)alloc(nH * 2);
  bf16_t* qrb  = (bf16_t*)alloc(nH * 2);
  bf16_t* kb   = (bf16_t*)alloc(nCat * 2);
  bf16_t* vtb  = (bf16_t*)alloc(nCat * 2);   // (B, DMODEL, KLEN) transposed V
  bf16_t* rhb  = (bf16_t*)alloc(nR * 2);
  float*  BD   = (float*)alloc((size_t)NHEADS * QLEN * KLEN * 4);  // 128 MB, reused per batch
  float*  avec = (float*)alloc(nH * 4);
  bf16_t* avb  = (bf16_t*)alloc(nH * 2);

  // prep
  k_build_cat<<<eblocks(nCat), 256, 0, stream>>>(mem, h, catb, hb);
  k_cast_bf16<<<eblocks(nR), 256, 0, stream>>>(r, rb, nR);
  {
    dim3 tb(32, 8);
    dim3 tgw(DMODEL / 32, DMODEL / 32, 1);
    k_transpose_cast<<<tgw, tb, 0, stream>>>(Wq, WqT, DMODEL, DMODEL);
    k_transpose_cast<<<tgw, tb, 0, stream>>>(Wk, WkT, DMODEL, DMODEL);
    k_transpose_cast<<<tgw, tb, 0, stream>>>(Wv, WvT, DMODEL, DMODEL);
    k_transpose_cast<<<tgw, tb, 0, stream>>>(Wr, WrT, DMODEL, DMODEL);
    k_transpose_cast<<<tgw, tb, 0, stream>>>(Wo, WoT, DMODEL, DMODEL);
  }

  auto gemm = [&](const bf16_t* A, const bf16_t* Bt, float* C, int ldc,
                  int M, int N, int K) {
    int tiles = (M / 16) * (N / 64);
    int blocks = (tiles + 7) / 8;
    gemm_bf16_wmma<DMODEL, DMODEL><<<blocks, 256, 0, stream>>>(A, Bt, C, ldc, M, N, K);
  };

  // projections
  gemm(hb,   WqT, qf,  DMODEL, BATCH * QLEN, DMODEL, DMODEL);
  gemm(catb, WkT, kf,  DMODEL, BATCH * KLEN, DMODEL, DMODEL);
  gemm(catb, WvT, vf,  DMODEL, BATCH * KLEN, DMODEL, DMODEL);
  gemm(rb,   WrT, rhf, DMODEL, KLEN,         DMODEL, DMODEL);

  // epilogues: biases + bf16 casts + V transpose
  k_bias_cast<<<eblocks(nH), 256, 0, stream>>>(qf, rwb, rrb, qwb, qrb);
  k_cast_bf16<<<eblocks(nCat), 256, 0, stream>>>(kf, kb, nCat);
  k_cast_bf16<<<eblocks(nR), 256, 0, stream>>>(rhf, rhb, nR);
  {
    dim3 tb(32, 8);
    dim3 tgv(KLEN / 32, DMODEL / 32, BATCH);
    k_transpose_cast<<<tgv, tb, 0, stream>>>(vf, vtb, KLEN, DMODEL);
  }

  // attention needs ~132 KB dynamic LDS (CDNA5 WGP has 320 KB)
  hipFuncSetAttribute(reinterpret_cast<const void*>(attn_kernel),
                      hipFuncAttributeMaxDynamicSharedMemorySize, ATT_LDS_BYTES);

  for (int b = 0; b < BATCH; ++b) {
    // BD[h] = qr(b,h) . r_head(h)^T   (M=1024, N=2048, K=64)
    for (int hh = 0; hh < NHEADS; ++hh)
      gemm(qrb + (size_t)b * QLEN * DMODEL + hh * DHEAD,
           rhb + hh * DHEAD,
           BD + (size_t)hh * QLEN * KLEN, KLEN,
           QLEN, KLEN, DHEAD);
    // fused scores + rel_shift + softmax + P@V
    attn_kernel<<<dim3(QLEN / 16, NHEADS), 256, ATT_LDS_BYTES, stream>>>(
        qwb, kb, vtb, BD, avec, b);
  }

  // output projection
  k_cast_bf16<<<eblocks(nH), 256, 0, stream>>>(avec, avb, nH);
  gemm(avb, WoT, out, DMODEL, BATCH * QLEN, DMODEL, DMODEL);

  (void)in_sizes; (void)n_in; (void)out_size; (void)ws_size;
}
